// MultiHeadAttention_67362267071076
// MI455X (gfx1250) — compile-verified, matrix operand order checked by reading, not run on hardware
//
#include <hip/hip_runtime.h>

#define D_MODEL 1024
#define N_HEADS 16
#define D_HEAD  64
#define BATCH   4
#define SEQ     2048
#define M_ROWS  (BATCH * SEQ)      // 8192
#define NEG_INF (-__builtin_inff())

typedef __attribute__((ext_vector_type(16))) _Float16 v16h;
typedef __attribute__((ext_vector_type(8)))  _Float16 v8h;
typedef __attribute__((ext_vector_type(8)))  float    v8f;
typedef __attribute__((ext_vector_type(4)))  float    v4f;

#define WMMA_F16(a, b, c) \
  __builtin_amdgcn_wmma_f32_16x16x32_f16(false, (a), false, (b), (short)0, (c), false, false)

// ---- WMMA fragment loaders (CDNA5 16-bit A/B layout, wave32) ----
// lane in [0,15]:  halves 0..7 = K k0..k0+7, halves 8..15 = K k0+16..k0+23
// lane in [16,31]: same rows/cols, K base shifted by +8 (hs=1)
__device__ __forceinline__ v16h frag_f16(const _Float16* base, int row, int ld, int k0, int hs) {
  const _Float16* p = base + (size_t)row * (size_t)ld + (size_t)(k0 + 8 * hs);
  v8h lo = *(const v8h*)(p);
  v8h hi = *(const v8h*)(p + 16);
  v16h r;
#pragma unroll
  for (int i = 0; i < 8; ++i) { r[i] = lo[i]; r[i + 8] = hi[i]; }
  return r;
}

// works for global f32 or LDS f32 (lowering to ds_load_b128 when base is LDS)
__device__ __forceinline__ v16h frag_f32(const float* base, int row, int ld, int k0, int hs) {
  const float* p = base + (size_t)row * (size_t)ld + (size_t)(k0 + 8 * hs);
  v4f a0 = *(const v4f*)(p);
  v4f a1 = *(const v4f*)(p + 4);
  v4f b0 = *(const v4f*)(p + 16);
  v4f b1 = *(const v4f*)(p + 20);
  v16h r;
#pragma unroll
  for (int i = 0; i < 4; ++i) {
    r[i]      = (_Float16)a0[i];
    r[i + 4]  = (_Float16)a1[i];
    r[i + 8]  = (_Float16)b0[i];
    r[i + 12] = (_Float16)b1[i];
  }
  return r;
}

// ---- 1. convert + transpose all 4 weight matrices to f16 [N,K] ----
__global__ void mha_wtrans(const float* __restrict__ wq, const float* __restrict__ wk,
                           const float* __restrict__ wv, const float* __restrict__ wfc,
                           _Float16* __restrict__ wqT, _Float16* __restrict__ wkT,
                           _Float16* __restrict__ wvT, _Float16* __restrict__ wfcT) {
  int i = blockIdx.x * blockDim.x + threadIdx.x;   // i = k*1024 + n
  int k = i >> 10, n = i & 1023;
  int t = n * D_MODEL + k;
  wqT[t]  = (_Float16)wq[i];
  wkT[t]  = (_Float16)wk[i];
  wvT[t]  = (_Float16)wv[i];
  wfcT[t] = (_Float16)wfc[i];
}

// ---- 2. QKV projection: [8192,1024]f32 @ Bt[1024,1024]f16 + bias -> f16 heads ----
// vmode==0: dst[b,h,l,d] ; vmode==1: dst[b,h,d,l] (V transposed for PV GEMM)
__global__ void mha_proj_gemm(const float* __restrict__ A, const _Float16* __restrict__ Bt,
                              const float* __restrict__ bias, _Float16* __restrict__ dst,
                              int vmode) {
  const int lane = threadIdx.x & 31;
  const int idx  = lane & 15;
  const int hs   = lane >> 4;
  const int wave = threadIdx.x >> 5;
  const int tile = blockIdx.x * 8 + wave;
  const int ntiles = D_MODEL / 16;                 // 64
  const int m0 = (tile / ntiles) * 16;
  const int n0 = (tile % ntiles) * 16;

  v8f c = {0.f, 0.f, 0.f, 0.f, 0.f, 0.f, 0.f, 0.f};
  for (int k0 = 0; k0 < D_MODEL; k0 += 32) {
    v16h a = frag_f32(A,  m0 + idx, D_MODEL, k0, hs);
    v16h b = frag_f16(Bt, n0 + idx, D_MODEL, k0, hs);
    c = WMMA_F16(a, b, c);
  }
#pragma unroll
  for (int r = 0; r < 8; ++r) {
    int m = m0 + r + 8 * hs;
    int n = n0 + idx;
    float val = c[r] + bias[n];
    int b = m >> 11, l = m & (SEQ - 1);
    int h = n >> 6,  d = n & 63;
    size_t o = vmode
        ? ((size_t)(b * N_HEADS + h) * D_HEAD + d) * SEQ + l
        : ((size_t)(b * N_HEADS + h) * SEQ + l) * D_HEAD + d;
    dst[o] = (_Float16)val;
  }
}

// ---- 3. fused attention: scores (WMMA) -> softmax (LDS) -> attn write -> PV (WMMA) ----
// One block per (head, 16-row strip). Score strip 16x2048 f32 lives in LDS (128KB,
// legal under CDNA5's 320KB/workgroup LDS). attn tensor is written exactly once.
__global__ void mha_fused_attn(const _Float16* __restrict__ Qh, const _Float16* __restrict__ Kh,
                               const _Float16* __restrict__ Vt, float* __restrict__ attn,
                               _Float16* __restrict__ Ch) {
  extern __shared__ float smem[];
  float* S       = smem;                       // [16][SEQ] score strip
  float* red     = S + 16 * SEQ;               // [256] partial reductions
  float* rowstat = red + 256;                  // [16] rowmax, then 1/rowsum
  float* pvbuf   = rowstat + 16;               // [4][16][16] PV cross-wave partials

  const int head = blockIdx.x >> 7;            // 64 heads
  const int l0   = (blockIdx.x & 127) * 16;    // strip start row
  const int tid  = threadIdx.x;
  const int wave = tid >> 5;
  const int lane = tid & 31;
  const int idx  = lane & 15;
  const int hs   = lane >> 4;

  const _Float16* Qb = Qh + (size_t)head * SEQ * D_HEAD;
  const _Float16* Kb = Kh + (size_t)head * SEQ * D_HEAD;
  const _Float16* vH = Vt + (size_t)head * D_HEAD * SEQ;

  // ---- phase 1: scores S = Q K^T / 8 for tiles on/below the diagonal ----
  v16h qa0 = frag_f16(Qb, l0 + idx, D_HEAD, 0,  hs);
  v16h qa1 = frag_f16(Qb, l0 + idx, D_HEAD, 32, hs);
  const int lastTile = l0 >> 4;                // tiles 0..l0/16 have valid columns
  for (int tt = wave; tt <= lastTile; tt += 8) {
    const int t0 = tt * 16;
    v8f c = {0.f, 0.f, 0.f, 0.f, 0.f, 0.f, 0.f, 0.f};
    v16h b0 = frag_f16(Kb, t0 + idx, D_HEAD, 0, hs);
    c = WMMA_F16(qa0, b0, c);
    v16h b1 = frag_f16(Kb, t0 + idx, D_HEAD, 32, hs);
    c = WMMA_F16(qa1, b1, c);
#pragma unroll
    for (int r = 0; r < 8; ++r) {
      int rr = r + 8 * hs;                     // strip row
      int t  = t0 + idx;
      S[rr * SEQ + t] = (t <= l0 + rr) ? c[r] * 0.125f : NEG_INF;
    }
  }
  __syncthreads();

  // ---- phase 2: row softmax over valid columns (16 threads per row) ----
  const int r   = tid >> 4;                    // strip row 0..15
  const int c16 = tid & 15;
  const int l   = l0 + r;
  const int len = l + 1;                       // valid columns per row

  float m = NEG_INF;
  for (int col = c16; col < len; col += 16) m = fmaxf(m, S[r * SEQ + col]);
  red[tid] = m;
  __syncthreads();
  if (c16 == 0) {
    float mm = red[r * 16];
#pragma unroll
    for (int i = 1; i < 16; ++i) mm = fmaxf(mm, red[r * 16 + i]);
    rowstat[r] = mm;
  }
  __syncthreads();
  const float rmax = rowstat[r];
  float s = 0.f;
  for (int col = c16; col < len; col += 16) {
    float e = __expf(S[r * SEQ + col] - rmax);
    S[r * SEQ + col] = e;
    s += e;
  }
  red[tid] = s;
  __syncthreads();
  if (c16 == 0) {
    float ss = 0.f;
#pragma unroll
    for (int i = 0; i < 16; ++i) ss += red[r * 16 + i];
    rowstat[r] = 1.f / ss;
  }
  __syncthreads();
  const float inv = rowstat[r];
  for (int col = c16; col < len; col += 16) S[r * SEQ + col] *= inv;
  {
    int col = l0 + c16;                        // zero masked part of diagonal tile
    if (col > l) S[r * SEQ + col] = 0.f;
    col = l0 + 16 + c16;                       // zero-pad to WMMA K granularity
    if (col < SEQ) S[r * SEQ + col] = 0.f;
  }
  __syncthreads();

  // ---- write normalized attn strip once (coalesced; zeros past the diagonal) ----
  float* aOut = attn + (size_t)head * SEQ * SEQ + (size_t)l0 * SEQ;
  for (int pos = tid; pos < 16 * SEQ; pos += 256) {
    int rr  = pos >> 11;
    int col = pos & (SEQ - 1);
    aOut[pos] = (col <= l0 + rr) ? S[rr * SEQ + col] : 0.f;
  }

  // ---- phase 3: O = P V from the LDS strip; 2 waves per d-tile split over K ----
  const int dt    = wave >> 1;                 // d-tile 0..3
  const int d0    = dt * 16;
  const int phase = wave & 1;
  const int kmax  = l0 + 16;                   // probs are zero past the diagonal
  v8f c = {0.f, 0.f, 0.f, 0.f, 0.f, 0.f, 0.f, 0.f};
  for (int k0 = phase * 32; k0 < kmax; k0 += 64) {
    v16h a = frag_f32(S, idx, SEQ, k0, hs);    // LDS -> ds_load_b128 fragments
    v16h b = frag_f16(vH, d0 + idx, SEQ, k0, hs);
    c = WMMA_F16(a, b, c);
  }
  if (phase == 1) {
#pragma unroll
    for (int rr = 0; rr < 8; ++rr) pvbuf[dt * 256 + (rr + 8 * hs) * 16 + idx] = c[rr];
  }
  __syncthreads();
  if (phase == 0) {
    const int bb = head >> 4, h = head & 15;
#pragma unroll
    for (int rr = 0; rr < 8; ++rr) {
      float val = c[rr] + pvbuf[dt * 256 + (rr + 8 * hs) * 16 + idx];
      int ll = l0 + rr + 8 * hs;
      int d  = d0 + idx;
      Ch[(size_t)(bb * SEQ + ll) * D_MODEL + h * D_HEAD + d] = (_Float16)val;
    }
  }
}

// ---- 4. output projection: Ch f16 @ wfcT f16 + bias -> f32 out ----
__global__ void mha_out_gemm(const _Float16* __restrict__ Ch, const _Float16* __restrict__ Bt,
                             const float* __restrict__ bias, float* __restrict__ out) {
  const int lane = threadIdx.x & 31;
  const int idx  = lane & 15;
  const int hs   = lane >> 4;
  const int wave = threadIdx.x >> 5;
  const int tile = blockIdx.x * 8 + wave;
  const int ntiles = D_MODEL / 16;
  const int m0 = (tile / ntiles) * 16;
  const int n0 = (tile % ntiles) * 16;

  v8f c = {0.f, 0.f, 0.f, 0.f, 0.f, 0.f, 0.f, 0.f};
  for (int k0 = 0; k0 < D_MODEL; k0 += 32) {
    v16h a = frag_f16(Ch, m0 + idx, D_MODEL, k0, hs);
    v16h b = frag_f16(Bt, n0 + idx, D_MODEL, k0, hs);
    c = WMMA_F16(a, b, c);
  }
#pragma unroll
  for (int r = 0; r < 8; ++r) {
    int m = m0 + r + 8 * hs;
    int n = n0 + idx;
    out[(size_t)m * D_MODEL + n] = c[r] + bias[n];
  }
}

extern "C" void kernel_launch(void* const* d_in, const int* in_sizes, int n_in,
                              void* d_out, int out_size, void* d_ws, size_t ws_size,
                              hipStream_t stream) {
  const float* query = (const float*)d_in[0];
  const float* key_  = (const float*)d_in[1];
  const float* value = (const float*)d_in[2];
  const float* w_q   = (const float*)d_in[3];
  const float* b_q   = (const float*)d_in[4];
  const float* w_k   = (const float*)d_in[5];
  const float* b_k   = (const float*)d_in[6];
  const float* w_v   = (const float*)d_in[7];
  const float* b_v   = (const float*)d_in[8];
  const float* w_fc  = (const float*)d_in[9];
  const float* b_fc  = (const float*)d_in[10];

  float* out  = (float*)d_out;                                   // [4,2048,1024]
  float* attn = out + (size_t)M_ROWS * D_MODEL;                  // [4,16,2048,2048]

  // workspace layout (f16 elements): 4 transposed weights + Qh + Kh + Vt + Ch
  _Float16* ws = (_Float16*)d_ws;
  const size_t WSZ  = (size_t)D_MODEL * D_MODEL;                 // 1M halves each
  const size_t QSZ  = (size_t)M_ROWS * D_MODEL;                  // 8M halves each
  _Float16* wqT  = ws;
  _Float16* wkT  = wqT + WSZ;
  _Float16* wvT  = wkT + WSZ;
  _Float16* wfcT = wvT + WSZ;
  _Float16* Qh   = wfcT + WSZ;
  _Float16* Kh   = Qh + QSZ;
  _Float16* Vt   = Kh + QSZ;
  _Float16* Ch   = Vt + QSZ;

  dim3 blk(256);

  mha_wtrans<<<(D_MODEL * D_MODEL) / 256, blk, 0, stream>>>(
      w_q, w_k, w_v, w_fc, wqT, wkT, wvT, wfcT);

  const int projTiles = (M_ROWS / 16) * (D_MODEL / 16);          // 32768
  mha_proj_gemm<<<projTiles / 8, blk, 0, stream>>>(query, wqT, b_q, Qh, 0);
  mha_proj_gemm<<<projTiles / 8, blk, 0, stream>>>(key_,  wkT, b_k, Kh, 0);
  mha_proj_gemm<<<projTiles / 8, blk, 0, stream>>>(value, wvT, b_v, Vt, 1);

  // fused scores+softmax+PV: one block per (head, 16-row strip)
  const size_t smemBytes = (size_t)(16 * SEQ + 256 + 16 + 4 * 256) * sizeof(float); // ~133KB
  mha_fused_attn<<<BATCH * N_HEADS * (SEQ / 16), blk, smemBytes, stream>>>(
      Qh, Kh, Vt, attn, Ch);

  mha_out_gemm<<<projTiles / 8, blk, 0, stream>>>(Ch, wfcT, b_fc, out);
}
